// TransformerEncoderLayer_CustomAttn_73031623901865
// MI455X (gfx1250) — matrix operand order repaired and checked by reading, not compiled, not run on hardware
//
#include <hip/hip_runtime.h>
#include <hip/hip_bf16.h>

typedef unsigned short u16;
typedef unsigned int   u32;

typedef __attribute__((ext_vector_type(16))) __bf16 v16bf;
typedef __attribute__((ext_vector_type(8)))  float  v8f;

union FragBF { v16bf v; u32 u[8]; };
union FragF  { v8f  v; float f[8]; };

#ifndef __has_builtin
#define __has_builtin(x) 0
#endif

#if __has_builtin(__builtin_amdgcn_global_load_async_to_lds_b128)
#define USE_ASYNC 1
#else
#define USE_ASYNC 0
#endif

// Pointee type of the async-load builtins encodes the transfer width:
// b128 -> GCC-vector int4, with AS1 (global) source and AS3 (LDS) dest.
typedef int v4i_b128 __attribute__((vector_size(16)));

// Copy 16 bytes global -> LDS. Async path uses the CDNA5 async-load engine
// (tracked by ASYNCcnt); fallback is a synchronous register round-trip with
// the same pipeline structure.
__device__ __forceinline__ void cp_b128(void* lds, const void* g) {
#if USE_ASYNC
  __builtin_amdgcn_global_load_async_to_lds_b128(
      (__attribute__((address_space(1))) v4i_b128*)g,
      (__attribute__((address_space(3))) v4i_b128*)lds, 0, 0);
#else
  *(uint4*)lds = *(const uint4*)g;
#endif
}
__device__ __forceinline__ void cp_wait() {
#if USE_ASYNC
#if __has_builtin(__builtin_amdgcn_s_wait_asynccnt)
  __builtin_amdgcn_s_wait_asynccnt(0);
#else
  asm volatile("s_wait_asynccnt 0x0" ::: "memory");
#endif
#endif
}

__device__ __forceinline__ u16 f2bf(float f) {
  u32 u = __float_as_uint(f);
  u += 0x7fffu + ((u >> 16) & 1u);  // round-to-nearest-even
  return (u16)(u >> 16);
}
__device__ __forceinline__ u32 pack2bf(float lo, float hi) {
  return (u32)f2bf(lo) | ((u32)f2bf(hi) << 16);
}

// ---- WMMA fragment loaders (CDNA5 ISA 7.12.2 layouts, bf16 16x16x32) ----
__device__ __forceinline__ v16bf frag_a(const u16* rowp, int lane, int kOff) {
  FragBF r;
  const int h8 = (lane & 16) ? 8 : 0;
#pragma unroll
  for (int i = 0; i < 8; ++i) {
    const int k = kOff + ((i & 3) << 1) + h8 + ((i & 4) ? 16 : 0);
    r.u[i] = *(const u32*)(rowp + k);
  }
  return r.v;
}
__device__ __forceinline__ v16bf frag_b(const u16* colp, int lane, int kOff) {
  FragBF r;
  const int h8 = (lane & 16) ? 8 : 0;
#pragma unroll
  for (int i = 0; i < 8; ++i) {
    const int k = kOff + ((i & 3) << 1) + h8 + ((i & 4) ? 16 : 0);
    r.u[i] = *(const u32*)(colp + k);
  }
  return r.v;
}
__device__ __forceinline__ v8f wmma_bf(v16bf a, v16bf b, v8f c) {
  return __builtin_amdgcn_wmma_f32_16x16x32_bf16(false, a, false, b, (short)0, c,
                                                 false, false);
}

// =====================================================================
// f32 -> bf16 bulk conversion (weights/activations pre-cast once).
// =====================================================================
__global__ __launch_bounds__(256) void f32_to_bf16_kernel(
    const float* __restrict__ in, u16* __restrict__ out, int n4) {
  const int i = blockIdx.x * 256 + threadIdx.x;
  if (i < n4) {
    const float4 f = ((const float4*)in)[i];
    uint2 p;
    p.x = pack2bf(f.x, f.y);
    p.y = pack2bf(f.z, f.w);
    ((uint2*)out)[i] = p;
  }
}

// =====================================================================
// Learnable positional embedding + L1 normalize.
// pe[b,q,s] = sum_p pos_w[q,p]*pos_embed[b,p,s] + pos_b[q]; L1-normalize over q,
// store transposed bf16: peT[(b*2048+s)*64 + q].
// =====================================================================
__global__ __launch_bounds__(256) void pos_pe_kernel(
    const float* __restrict__ pos_embed, const float* __restrict__ pw,
    const float* __restrict__ pb, u16* __restrict__ peT) {
  __shared__ float Wsh[64 * 64];
  __shared__ float Bsh[64];
  const int tid = threadIdx.x;
#pragma unroll
  for (int ii = 0; ii < 16; ++ii) Wsh[tid + ii * 256] = pw[tid + ii * 256];
  if (tid < 64) Bsh[tid] = pb[tid];
  __syncthreads();

  const int g = blockIdx.x * 256 + tid;  // g = b*2048 + s
  const int b = g >> 11, s = g & 2047;
  float col[64];
#pragma unroll
  for (int p = 0; p < 64; ++p)
    col[p] = pos_embed[(size_t)b * 64 * 2048 + (size_t)p * 2048 + s];

  float l1 = 0.f;
  for (int q = 0; q < 64; ++q) {
    float a = Bsh[q];
#pragma unroll
    for (int p = 0; p < 64; ++p) a += Wsh[q * 64 + p] * col[p];
    l1 += fabsf(a);
  }
  const float inv = 1.0f / fmaxf(l1, 1e-12f);
  for (int q = 0; q < 64; ++q) {
    float a = Bsh[q];
#pragma unroll
    for (int p = 0; p < 64; ++p) a += Wsh[q * 64 + p] * col[p];
    peT[((size_t)g << 6) + q] = f2bf(a * inv);
  }
}

// =====================================================================
// bf16-WMMA GEMM: C[M,N] = A[M,K] * W[N,K]^T (+bias, epilogue), A & W bf16.
// Block tile 128x64, BK=64, 256 threads = 8 waves (4x2), each wave 32x32.
// Double-buffered LDS fed by async global->LDS copies (overlaps with WMMA).
// MODE 0: QKV scatter (+ q scaling). MODE 1: resid + acc + bias -> f32.
// MODE 2: relu(acc+bias) -> bf16.
// =====================================================================
template <int MODE>
__global__ __launch_bounds__(256) void gemm_bf16_kernel(
    const u16* __restrict__ A, const u16* __restrict__ W,
    const float* __restrict__ bias, const float* __restrict__ resid,
    float* __restrict__ outF, u16* __restrict__ outB, u16* __restrict__ qO,
    u16* __restrict__ kO, u16* __restrict__ vO, int M, int N, int K) {
  __shared__ u16 As[2][128 * 64];  // 16KB x2
  __shared__ u16 Ws[2][64 * 64];   //  8KB x2
  const int tid = threadIdx.x, lane = tid & 31, wid = tid >> 5;
  const int wm = wid & 3, wn = wid >> 2;
  const int r0 = blockIdx.x * 128, n0 = blockIdx.y * 64;

  FragF acc[2][2];
#pragma unroll
  for (int a = 0; a < 2; ++a)
#pragma unroll
    for (int bb = 0; bb < 2; ++bb)
#pragma unroll
      for (int e = 0; e < 8; ++e) acc[a][bb].f[e] = 0.f;

  auto issue = [&](int buf, int k0) {
#pragma unroll
    for (int ii = 0; ii < 4; ++ii) {  // 1024 chunks of 8 bf16 (A: 128x64)
      const int c = tid + ii * 256;
      const int r = c >> 3, c8 = (c & 7) << 3;
      cp_b128(&As[buf][r * 64 + c8], A + (size_t)(r0 + r) * K + k0 + c8);
    }
#pragma unroll
    for (int ii = 0; ii < 2; ++ii) {  // 512 chunks (W: 64x64)
      const int c = tid + ii * 256;
      const int r = c >> 3, c8 = (c & 7) << 3;
      cp_b128(&Ws[buf][r * 64 + c8], W + (size_t)(n0 + r) * K + k0 + c8);
    }
  };

  issue(0, 0);
  const int T = K >> 6;
  for (int it = 0; it < T; ++it) {
    cp_wait();
    __syncthreads();  // tile `it` resident; prev buffer free for refill
    if (it + 1 < T) issue((it + 1) & 1, (it + 1) << 6);
    const int buf = it & 1;
#pragma unroll
    for (int ks = 0; ks < 2; ++ks) {
      const int ko = ks << 5;
      const v16bf a0 = frag_a(&As[buf][(wm * 32 + (lane & 15)) * 64], lane, ko);
      const v16bf a1 =
          frag_a(&As[buf][(wm * 32 + 16 + (lane & 15)) * 64], lane, ko);
      const v16bf b0 = frag_b(&Ws[buf][(wn * 32 + (lane & 15)) * 64], lane, ko);
      const v16bf b1 =
          frag_b(&Ws[buf][(wn * 32 + 16 + (lane & 15)) * 64], lane, ko);
      acc[0][0].v = wmma_bf(a0, b0, acc[0][0].v);
      acc[0][1].v = wmma_bf(a0, b1, acc[0][1].v);
      acc[1][0].v = wmma_bf(a1, b0, acc[1][0].v);
      acc[1][1].v = wmma_bf(a1, b1, acc[1][1].v);
    }
  }

#pragma unroll
  for (int fn = 0; fn < 2; ++fn) {
    const int n = n0 + wn * 32 + fn * 16 + (lane & 15);
    const float bv = bias[n];
#pragma unroll
    for (int fm = 0; fm < 2; ++fm) {
#pragma unroll
      for (int e = 0; e < 8; ++e) {
        const int m = r0 + wm * 32 + fm * 16 + e + ((lane & 16) ? 8 : 0);
        float val = acc[fm][fn].f[e] + bv;
        if (MODE == 0) {
          const int s = m >> 1, bb = m & 1;
          const int which = n >> 10, ee = n & 1023, h = ee >> 6, d = ee & 63;
          if (which == 0) val *= 0.125f;  // Dh^-0.5
          u16* dst = (which == 0) ? qO : ((which == 1) ? kO : vO);
          dst[((size_t)((bb * 16 + h) * 2048 + s) << 6) + d] = f2bf(val);
        } else if (MODE == 1) {
          const size_t idx = (size_t)m * N + n;
          outF[idx] = resid[idx] + val;
        } else {
          outB[(size_t)m * N + n] = f2bf(fmaxf(val, 0.0f));
        }
      }
    }
  }
}

// =====================================================================
// Flash attention with fused positional-similarity (extra K=64 via peT).
// Grid: (S/128, B*H). 8 waves, each owns 16 query rows; loops key blocks of 64.
// =====================================================================
__global__ __launch_bounds__(256) void flash_attn_kernel(
    const u16* __restrict__ qp, const u16* __restrict__ kp,
    const u16* __restrict__ vp, const u16* __restrict__ pep,
    u16* __restrict__ ctx) {
  __shared__ u16 Ks[64 * 72];      // [t][d]
  __shared__ u16 PEs[64 * 72];     // [t][p]
  __shared__ u16 Vt[64 * 72];      // [d][t]  (transposed V)
  __shared__ u16 Pw[8 * 16 * 72];  // per-wave P scratch [row][t]

  const int tid = threadIdx.x, lane = tid & 31, wid = tid >> 5;
  const int bh = blockIdx.y, b = bh >> 4, h = bh & 15;
  const size_t headBase = (size_t)bh << 17;  // *2048*64
  const size_t peBase = (size_t)b << 17;
  const int s0 = blockIdx.x * 128;
  const int sRowA = s0 + wid * 16 + (lane & 15);

  FragBF aq[2], ape[2];
  {
    const u16* qrow = qp + headBase + ((size_t)sRowA << 6);
    const u16* prow = pep + peBase + ((size_t)sRowA << 6);
    aq[0].v = frag_a(qrow, lane, 0);
    aq[1].v = frag_a(qrow, lane, 32);
    ape[0].v = frag_a(prow, lane, 0);
    ape[1].v = frag_a(prow, lane, 32);
  }

  float mrun[8], lrun[8];
  FragF oacc[4];
#pragma unroll
  for (int e = 0; e < 8; ++e) { mrun[e] = -3.0e38f; lrun[e] = 0.f; }
#pragma unroll
  for (int jd = 0; jd < 4; ++jd)
#pragma unroll
    for (int e = 0; e < 8; ++e) oacc[jd].f[e] = 0.f;

  const float L2E = 1.44269504f;

  for (int kt0 = 0; kt0 < 2048; kt0 += 64) {
    __syncthreads();
    // K and PE tiles via async copy engine (8 bf16 = 16B chunks)
#pragma unroll
    for (int ii = 0; ii < 2; ++ii) {
      const int c = tid + ii * 256;  // 512 chunks
      const int t = c >> 3, c8 = (c & 7) << 3;
      const size_t gk = headBase + ((size_t)(kt0 + t) << 6) + c8;
      cp_b128(&Ks[t * 72 + c8], kp + gk);
      cp_b128(&PEs[t * 72 + c8],
              pep + peBase + ((size_t)(kt0 + t) << 6) + c8);
    }
    // V tile loaded synchronously with transpose-store (overlaps with async)
#pragma unroll
    for (int ii = 0; ii < 4; ++ii) {
      const int i = tid + ii * 256;
      const int t = i >> 4, d4 = (i & 15) << 2;
      const uint2 vv =
          *(const uint2*)(vp + headBase + ((size_t)(kt0 + t) << 6) + d4);
      Vt[(d4 + 0) * 72 + t] = (u16)(vv.x & 0xffffu);
      Vt[(d4 + 1) * 72 + t] = (u16)(vv.x >> 16);
      Vt[(d4 + 2) * 72 + t] = (u16)(vv.y & 0xffffu);
      Vt[(d4 + 3) * 72 + t] = (u16)(vv.y >> 16);
    }
    cp_wait();
    __syncthreads();

    FragF sc[4];
#pragma unroll
    for (int jn = 0; jn < 4; ++jn) {
      const u16* kcol = &Ks[(jn * 16 + (lane & 15)) * 72];
      const u16* pcol = &PEs[(jn * 16 + (lane & 15)) * 72];
      v8f a = {0.f, 0.f, 0.f, 0.f, 0.f, 0.f, 0.f, 0.f};
      a = wmma_bf(aq[0].v, frag_b(kcol, lane, 0), a);
      a = wmma_bf(aq[1].v, frag_b(kcol, lane, 32), a);
      a = wmma_bf(ape[0].v, frag_b(pcol, lane, 0), a);  // + pos_sim
      a = wmma_bf(ape[1].v, frag_b(pcol, lane, 32), a);
      sc[jn].v = a;
    }

    // online softmax per row (row = e + 8*(lane>=16); stats over 16 lanes)
#pragma unroll
    for (int e = 0; e < 8; ++e) {
      float rm = fmaxf(fmaxf(sc[0].f[e], sc[1].f[e]),
                       fmaxf(sc[2].f[e], sc[3].f[e]));
      rm = fmaxf(rm, __shfl_xor(rm, 1, 32));
      rm = fmaxf(rm, __shfl_xor(rm, 2, 32));
      rm = fmaxf(rm, __shfl_xor(rm, 4, 32));
      rm = fmaxf(rm, __shfl_xor(rm, 8, 32));
      const float mnew = fmaxf(mrun[e], rm);
      const float scale = exp2f((mrun[e] - mnew) * L2E);
      float rs = 0.f;
#pragma unroll
      for (int jn = 0; jn < 4; ++jn) {
        const float p = exp2f((sc[jn].f[e] - mnew) * L2E);
        sc[jn].f[e] = p;
        rs += p;
      }
      rs += __shfl_xor(rs, 1, 32);
      rs += __shfl_xor(rs, 2, 32);
      rs += __shfl_xor(rs, 4, 32);
      rs += __shfl_xor(rs, 8, 32);
      lrun[e] = lrun[e] * scale + rs;
      mrun[e] = mnew;
#pragma unroll
      for (int jd = 0; jd < 4; ++jd) oacc[jd].f[e] *= scale;
    }

    // P (C layout) -> wave-private LDS -> A-fragment layout
    u16* pw = &Pw[wid * (16 * 72)];
    const int rb = (lane & 16) ? 8 : 0;
#pragma unroll
    for (int e = 0; e < 8; ++e)
#pragma unroll
      for (int jn = 0; jn < 4; ++jn)
        pw[(e + rb) * 72 + jn * 16 + (lane & 15)] = f2bf(sc[jn].f[e]);

    const u16* prow = &pw[(lane & 15) * 72];
    const v16bf ap0 = frag_a(prow, lane, 0);
    const v16bf ap1 = frag_a(prow, lane, 32);
#pragma unroll
    for (int jd = 0; jd < 4; ++jd) {
      const u16* vcol = &Vt[(jd * 16 + (lane & 15)) * 72];
      oacc[jd].v = wmma_bf(ap0, frag_b(vcol, lane, 0), oacc[jd].v);
      oacc[jd].v = wmma_bf(ap1, frag_b(vcol, lane, 32), oacc[jd].v);
    }
  }

  // write ctx as bf16 in GEMM-A layout: [(s*B+b), h*64+d]
#pragma unroll
  for (int e = 0; e < 8; ++e) {
    const float inv = 1.0f / lrun[e];
    const int srow = s0 + wid * 16 + e + ((lane & 16) ? 8 : 0);
#pragma unroll
    for (int jd = 0; jd < 4; ++jd) {
      const int d = jd * 16 + (lane & 15);
      ctx[((size_t)(srow * 2 + b) << 10) + h * 64 + d] =
          f2bf(oacc[jd].f[e] * inv);
    }
  }
}

// =====================================================================
// LayerNorm over last dim (1024). One block per row.
// =====================================================================
__global__ __launch_bounds__(256) void layernorm_kernel(
    const float* __restrict__ in, const float* __restrict__ g,
    const float* __restrict__ be, float* __restrict__ outF,
    u16* __restrict__ outB) {
  __shared__ float red[256];
  const int tid = threadIdx.x;
  const size_t base = (size_t)blockIdx.x << 10;
  float x[4];
#pragma unroll
  for (int j = 0; j < 4; ++j) x[j] = in[base + tid + j * 256];

  float s = x[0] + x[1] + x[2] + x[3];
  red[tid] = s;
  __syncthreads();
  for (int o = 128; o > 0; o >>= 1) {
    if (tid < o) red[tid] += red[tid + o];
    __syncthreads();
  }
  const float mu = red[0] * (1.0f / 1024.0f);
  __syncthreads();

  float s2 = 0.f;
#pragma unroll
  for (int j = 0; j < 4; ++j) {
    const float d = x[j] - mu;
    s2 += d * d;
  }
  red[tid] = s2;
  __syncthreads();
  for (int o = 128; o > 0; o >>= 1) {
    if (tid < o) red[tid] += red[tid + o];
    __syncthreads();
  }
  const float rstd = rsqrtf(red[0] * (1.0f / 1024.0f) + 1e-5f);

#pragma unroll
  for (int j = 0; j < 4; ++j) {
    const int e = tid + j * 256;
    const float y = (x[j] - mu) * rstd * g[e] + be[e];
    outF[base + e] = y;
    if (outB) outB[base + e] = f2bf(y);
  }
}

// =====================================================================
extern "C" void kernel_launch(void* const* d_in, const int* in_sizes, int n_in,
                              void* d_out, int out_size, void* d_ws,
                              size_t ws_size, hipStream_t stream) {
  (void)in_sizes; (void)n_in; (void)out_size; (void)ws_size;
  const float* src       = (const float*)d_in[0];
  const float* pos_embed = (const float*)d_in[1];
  const float* in_proj_w = (const float*)d_in[2];
  const float* in_proj_b = (const float*)d_in[3];
  const float* out_w     = (const float*)d_in[4];
  const float* out_b     = (const float*)d_in[5];
  const float* lin1_w    = (const float*)d_in[6];
  const float* lin1_b    = (const float*)d_in[7];
  const float* lin2_w    = (const float*)d_in[8];
  const float* lin2_b    = (const float*)d_in[9];
  const float* ln1_g     = (const float*)d_in[10];
  const float* ln1_bb    = (const float*)d_in[11];
  const float* ln2_g     = (const float*)d_in[12];
  const float* ln2_bb    = (const float*)d_in[13];
  const float* pos_w     = (const float*)d_in[14];
  const float* pos_b     = (const float*)d_in[15];
  float* out = (float*)d_out;

  constexpr size_t M = 4096;  // S*B rows
  char* ws = (char*)d_ws;
  size_t off = 0;
  auto take = [&](size_t bytes) {
    void* p = ws + off;
    off = (off + bytes + 255) & ~(size_t)255;
    return p;
  };
  u16*   qB   = (u16*)take(2ull * 16 * 2048 * 64 * 2);  // [B,H,S,Dh] bf16
  u16*   kB   = (u16*)take(2ull * 16 * 2048 * 64 * 2);
  u16*   vB   = (u16*)take(2ull * 16 * 2048 * 64 * 2);
  u16*   peT  = (u16*)take(2ull * 2048 * 64 * 2);       // [B,S,P] bf16
  u16*   ctxB = (u16*)take(M * 1024 * 2);
  float* y1   = (float*)take(M * 1024 * 4);
  float* xF   = (float*)take(M * 1024 * 4);
  u16*   xB   = (u16*)take(M * 1024 * 2);
  u16*   hB   = (u16*)take(M * 4096 * 2);
  float* y2   = (float*)take(M * 1024 * 4);
  u16*   srcB = (u16*)take(M * 1024 * 2);               // bf16 pre-casts
  u16*   ipwB = (u16*)take(3072ull * 1024 * 2);
  u16*   owB  = (u16*)take(1024ull * 1024 * 2);
  u16*   l1wB = (u16*)take(4096ull * 1024 * 2);
  u16*   l2wB = (u16*)take(1024ull * 4096 * 2);

  // 0) one-time bf16 pre-casts (amortized across all GEMM block rows)
  f32_to_bf16_kernel<<<4096, 256, 0, stream>>>(src, srcB, 1048576);
  f32_to_bf16_kernel<<<3072, 256, 0, stream>>>(in_proj_w, ipwB, 786432);
  f32_to_bf16_kernel<<<1024, 256, 0, stream>>>(out_w, owB, 262144);
  f32_to_bf16_kernel<<<4096, 256, 0, stream>>>(lin1_w, l1wB, 1048576);
  f32_to_bf16_kernel<<<4096, 256, 0, stream>>>(lin2_w, l2wB, 1048576);

  // 1) positional embedding + L1 normalize
  pos_pe_kernel<<<16, 256, 0, stream>>>(pos_embed, pos_w, pos_b, peT);

  // 2) QKV projection: [4096,1024] x [3072,1024]^T, scatter to heads (q scaled)
  gemm_bf16_kernel<0><<<dim3(32, 48), 256, 0, stream>>>(
      srcB, ipwB, in_proj_b, nullptr, nullptr, nullptr, qB, kB, vB,
      4096, 3072, 1024);

  // 3) flash attention with fused pos_sim
  flash_attn_kernel<<<dim3(16, 32), 256, 0, stream>>>(qB, kB, vB, peT, ctxB);

  // 4) out projection + residual: y1 = src + ctx @ out_w^T + out_b
  gemm_bf16_kernel<1><<<dim3(32, 16), 256, 0, stream>>>(
      ctxB, owB, out_b, src, y1, nullptr, nullptr, nullptr, nullptr,
      4096, 1024, 1024);

  // 5) LN1 -> x (f32 + bf16)
  layernorm_kernel<<<4096, 256, 0, stream>>>(y1, ln1_g, ln1_bb, xF, xB);

  // 6) FFN up: h = relu(x @ lin1^T + b1) (bf16)
  gemm_bf16_kernel<2><<<dim3(32, 64), 256, 0, stream>>>(
      xB, l1wB, lin1_b, nullptr, nullptr, hB, nullptr, nullptr, nullptr,
      4096, 4096, 1024);

  // 7) FFN down + residual: y2 = x + h @ lin2^T + b2
  gemm_bf16_kernel<1><<<dim3(32, 16), 256, 0, stream>>>(
      hB, l2wB, lin2_b, xF, y2, nullptr, nullptr, nullptr, nullptr,
      4096, 1024, 4096);

  // 8) LN2 -> output
  layernorm_kernel<<<4096, 256, 0, stream>>>(y2, ln2_g, ln2_bb, out, nullptr);
}
